// LongRetrievalModel_47717086659263
// MI455X (gfx1250) — compile-verified, matrix operand order checked by reading, not run on hardware
//
#include <hip/hip_runtime.h>
#include <hip/hip_bf16.h>
#include <math.h>

#define BB 4
#define SS 2048
#define DD 384
#define HH 12
#define DKK 32
#define LLAY 6
#define FFD 1536
#define MM (BB*SS)
#define PP 512
#define NQKV (3*DD)   // 1152

typedef __attribute__((ext_vector_type(16))) __bf16 v16bf;
typedef __attribute__((ext_vector_type(8)))  float  v8f;

union FragU { v16bf v; uint4 u[2]; };

// ---------------------------------------------------------------------------
// Block reduction helper (blockDim.x == 384, padded to 512)
// ---------------------------------------------------------------------------
__device__ __forceinline__ float block_sum_384(float v, float* red) {
  int tid = threadIdx.x;
  red[tid] = v;
  if (tid < 128) red[384 + tid] = 0.f;
  __syncthreads();
  for (int s = 256; s > 0; s >>= 1) {
    if (tid < s) red[tid] += red[tid + s];
    __syncthreads();
  }
  float r = red[0];
  __syncthreads();
  return r;
}

// ---------------------------------------------------------------------------
// Embedding + positional interpolation + token-type + LayerNorm
// ---------------------------------------------------------------------------
__global__ void embed_ln_kernel(const int* __restrict__ ids,
                                const float* __restrict__ etok,
                                const float* __restrict__ epos,
                                const float* __restrict__ ett,
                                const float* __restrict__ g,
                                const float* __restrict__ b,
                                float* __restrict__ hid,
                                __bf16* __restrict__ hidb) {
  int row = blockIdx.x;
  int d   = threadIdx.x;
  int s   = row % SS;
  int id  = ids[row];
  const float scale = (float)(PP - 1) / (float)(SS - 1);
  float p  = (float)s * scale;
  int lo   = (int)floorf(p);
  int hi   = min(lo + 1, PP - 1);
  float w  = p - (float)lo;
  float val = etok[(size_t)id * DD + d]
            + (1.f - w) * epos[(size_t)lo * DD + d]
            + w * epos[(size_t)hi * DD + d]
            + ett[d];
  __shared__ float red[512];
  float mean = block_sum_384(val, red) * (1.f / DD);
  float c    = val - mean;
  float var  = block_sum_384(c * c, red) * (1.f / DD);
  float y    = c * rsqrtf(var + 1e-5f) * g[d] + b[d];
  hid[(size_t)row * DD + d]  = y;
  hidb[(size_t)row * DD + d] = (__bf16)y;
}

// ---------------------------------------------------------------------------
// Residual add + LayerNorm
// ---------------------------------------------------------------------------
__global__ void add_ln_kernel(float* __restrict__ hid,
                              const float* __restrict__ x,
                              const float* __restrict__ g,
                              const float* __restrict__ b,
                              __bf16* __restrict__ hidb) {
  int row = blockIdx.x;
  int d   = threadIdx.x;
  size_t idx = (size_t)row * DD + d;
  float v = hid[idx] + x[idx];
  __shared__ float red[512];
  float mean = block_sum_384(v, red) * (1.f / DD);
  float c    = v - mean;
  float var  = block_sum_384(c * c, red) * (1.f / DD);
  float y    = c * rsqrtf(var + 1e-5f) * g[d] + b[d];
  hid[idx]  = y;
  hidb[idx] = (__bf16)y;
}

// ---------------------------------------------------------------------------
// Weight convert + transpose: W (K x Nw f32 row-major) -> Wt (Nw x K bf16)
// ---------------------------------------------------------------------------
__global__ void conv_w_kernel(const float* __restrict__ W, __bf16* __restrict__ Wt,
                              int K, int N) {
  int idx = blockIdx.x * blockDim.x + threadIdx.x;
  if (idx >= K * N) return;
  int k = idx / N, n = idx % N;
  Wt[(size_t)n * K + k] = (__bf16)W[idx];
}

// pack q/k/v biases into one 1152-entry vector
__global__ void pack_bias_kernel(const float* __restrict__ bq,
                                 const float* __restrict__ bk,
                                 const float* __restrict__ bv,
                                 float* __restrict__ out) {
  int i = blockIdx.x * blockDim.x + threadIdx.x;
  if (i >= NQKV) return;
  out[i] = (i < DD) ? bq[i] : (i < 2 * DD) ? bk[i - DD] : bv[i - 2 * DD];
}

// ---------------------------------------------------------------------------
// WMMA bf16 GEMM, 64x64 block tile, 4 waves (2x2), 32x32 register tile/wave.
// 2 A frags x 2 B frags -> 4 v_wmma_f32_16x16x32_bf16 per K-step.
// MODE 0: C f32 = acc + bias          MODE 1: C bf16 = gelu(acc + bias)
// ---------------------------------------------------------------------------
template <int MODE>
__global__ void gemm_bf16_kernel(const __bf16* __restrict__ A,
                                 const __bf16* __restrict__ Wt,
                                 const float* __restrict__ bias,
                                 void* __restrict__ Cout,
                                 int M, int N, int K) {
  const int lane  = threadIdx.x & 31;
  const int wave  = threadIdx.x >> 5;
  const int waveM = wave >> 1;          // 0..1
  const int waveN = wave & 1;           // 0..1
  const int m0    = blockIdx.y * 64 + waveM * 32;   // 32 rows per wave
  const int n0    = blockIdx.x * 64 + waveN * 32;   // 32 cols per wave
  const int half  = lane >> 4;
  const int l15   = lane & 15;

  const __bf16* arow0 = A  + (size_t)(m0 + l15) * K + half * 8;
  const __bf16* arow1 = arow0 + (size_t)16 * K;
  const __bf16* brow0 = Wt + (size_t)(n0 + l15) * K + half * 8;
  const __bf16* brow1 = brow0 + (size_t)16 * K;

  v8f acc00 = {}, acc01 = {}, acc10 = {}, acc11 = {};
  for (int k0 = 0; k0 < K; k0 += 32) {
    FragU a0, a1, b0, b1;
    a0.u[0] = *(const uint4*)(arow0 + k0);
    a0.u[1] = *(const uint4*)(arow0 + k0 + 16);
    a1.u[0] = *(const uint4*)(arow1 + k0);
    a1.u[1] = *(const uint4*)(arow1 + k0 + 16);
    b0.u[0] = *(const uint4*)(brow0 + k0);
    b0.u[1] = *(const uint4*)(brow0 + k0 + 16);
    b1.u[0] = *(const uint4*)(brow1 + k0);
    b1.u[1] = *(const uint4*)(brow1 + k0 + 16);
    __builtin_prefetch(arow0 + k0 + 64, 0, 0);   // global_prefetch_b8
    __builtin_prefetch(brow0 + k0 + 64, 0, 0);
    acc00 = __builtin_amdgcn_wmma_f32_16x16x32_bf16(false, a0.v, false, b0.v,
                                                    (short)0, acc00, false, false);
    acc01 = __builtin_amdgcn_wmma_f32_16x16x32_bf16(false, a0.v, false, b1.v,
                                                    (short)0, acc01, false, false);
    acc10 = __builtin_amdgcn_wmma_f32_16x16x32_bf16(false, a1.v, false, b0.v,
                                                    (short)0, acc10, false, false);
    acc11 = __builtin_amdgcn_wmma_f32_16x16x32_bf16(false, a1.v, false, b1.v,
                                                    (short)0, acc11, false, false);
  }

  v8f accs[2][2] = {{acc00, acc01}, {acc10, acc11}};
  #pragma unroll
  for (int i = 0; i < 2; ++i) {
    #pragma unroll
    for (int j = 0; j < 2; ++j) {
      #pragma unroll
      for (int r = 0; r < 8; ++r) {
        int row = m0 + i * 16 + r + half * 8;
        int col = n0 + j * 16 + l15;
        float val = accs[i][j][r] + bias[col];
        if (MODE == 0) {
          ((float*)Cout)[(size_t)row * N + col] = val;
        } else {
          float ge = 0.5f * val * (1.f + erff(val * 0.70710678118654752f));
          ((__bf16*)Cout)[(size_t)row * N + col] = (__bf16)ge;
        }
      }
    }
  }
}

// ---------------------------------------------------------------------------
// beta = sigmoid(hidden @ Wbeta + bbeta) * mask   (N = H = 12, scalar path)
// ---------------------------------------------------------------------------
__global__ void beta_kernel(const float* __restrict__ hid,
                            const float* __restrict__ Wb,
                            const float* __restrict__ bb,
                            const int* __restrict__ mask,
                            float* __restrict__ beta) {
  int idx = blockIdx.x * blockDim.x + threadIdx.x;
  if (idx >= MM * HH) return;
  int r = idx / HH, h = idx % HH;
  float acc = bb[h];
  const float* hr = hid + (size_t)r * DD;
  #pragma unroll 4
  for (int d = 0; d < DD; ++d) acc += hr[d] * Wb[(size_t)d * HH + h];
  float sg = 1.f / (1.f + expf(-acc));
  beta[idx] = sg * (float)mask[r];
}

// ---------------------------------------------------------------------------
// Delta-rule recurrence over fused qkv rows (stride 1152: q|k|v).
// One block per (b,h); wave0 = fast decay, wave1 = slow decay.
// Lane j holds state column S[:,j] in 32 f32 registers.
// ---------------------------------------------------------------------------
__global__ void delta_kernel(const float* __restrict__ qkv,
                             const float* __restrict__ beta,
                             const float* __restrict__ fdec,
                             const float* __restrict__ sdec,
                             float* __restrict__ o2) {
  int bh = blockIdx.x;                 // 0 .. B*H-1
  int b  = bh / HH, h = bh % HH;
  int wave = threadIdx.x >> 5;
  int lane = threadIdx.x & 31;
  float dec = (wave == 0) ? fdec[h] : sdec[h];
  float* obuf = o2 + (size_t)wave * MM * DD;

  float st[DKK];
  #pragma unroll
  for (int i = 0; i < DKK; ++i) st[i] = 0.f;

  for (int t = 0; t < SS; ++t) {
    const float* rowp = qkv + (size_t)(b * SS + t) * NQKV + h * DKK + lane;
    float qv = rowp[0];
    float kv = rowp[DD];
    float vv = rowp[2 * DD];
    float bt = beta[(size_t)(b * SS + t) * HH + h];

    float kk[DKK];
    #pragma unroll
    for (int i = 0; i < DKK; ++i) kk[i] = __shfl(kv, i, 32);

    float pred = 0.f;
    #pragma unroll
    for (int i = 0; i < DKK; ++i) { st[i] *= dec; pred += kk[i] * st[i]; }

    float dlt = bt * (vv - pred);
    #pragma unroll
    for (int i = 0; i < DKK; ++i) st[i] += kk[i] * dlt;

    float out = 0.f;
    #pragma unroll
    for (int i = 0; i < DKK; ++i) out += __shfl(qv, i, 32) * st[i];

    obuf[(size_t)(b * SS + t) * DD + h * DKK + lane] = out;
  }
}

// o_bf = bf16(o_fast + o_slow)
__global__ void combine_o_kernel(const float* __restrict__ o2,
                                 __bf16* __restrict__ y, int n) {
  int i = blockIdx.x * blockDim.x + threadIdx.x;
  if (i >= n) return;
  y[i] = (__bf16)(o2[i] + o2[(size_t)n + i]);
}

// ---------------------------------------------------------------------------
// Masked mean pool + L2 normalize
// ---------------------------------------------------------------------------
__global__ void pool_kernel(const float* __restrict__ hid,
                            const int* __restrict__ mask,
                            float* __restrict__ out) {
  int b = blockIdx.x, d = threadIdx.x;
  float acc = 0.f, cnt = 0.f;
  for (int s = 0; s < SS; ++s) {
    float m = (float)mask[b * SS + s];
    acc += hid[((size_t)b * SS + s) * DD + d] * m;
    cnt += m;
  }
  float pooled = acc / fmaxf(cnt, 1e-9f);
  __shared__ float red[512];
  float sq = block_sum_384(pooled * pooled, red);
  out[(size_t)b * DD + d] = pooled / fmaxf(sqrtf(sq), 1e-12f);
}

// ---------------------------------------------------------------------------
// Launch
// ---------------------------------------------------------------------------
extern "C" void kernel_launch(void* const* d_in, const int* in_sizes, int n_in,
                              void* d_out, int out_size, void* d_ws, size_t ws_size,
                              hipStream_t stream) {
  const int*   ids     = (const int*)d_in[0];
  const int*   mask    = (const int*)d_in[1];
  const float* emb_tok = (const float*)d_in[2];
  const float* emb_pos = (const float*)d_in[3];
  const float* emb_tt  = (const float*)d_in[4];
  const float* en_g    = (const float*)d_in[5];
  const float* en_b    = (const float*)d_in[6];
  const float* Wq      = (const float*)d_in[7];
  const float* bq      = (const float*)d_in[8];
  const float* Wk      = (const float*)d_in[9];
  const float* bk      = (const float*)d_in[10];
  const float* Wv      = (const float*)d_in[11];
  const float* bv      = (const float*)d_in[12];
  const float* Wbeta   = (const float*)d_in[13];
  const float* bbeta   = (const float*)d_in[14];
  const float* fdec    = (const float*)d_in[15];
  const float* sdec    = (const float*)d_in[16];
  const float* Wo      = (const float*)d_in[17];
  const float* bo      = (const float*)d_in[18];
  const float* n1_g    = (const float*)d_in[19];
  const float* n1_b    = (const float*)d_in[20];
  const float* W1      = (const float*)d_in[21];
  const float* b1      = (const float*)d_in[22];
  const float* W2      = (const float*)d_in[23];
  const float* b2      = (const float*)d_in[24];
  const float* n2_g    = (const float*)d_in[25];
  const float* n2_b    = (const float*)d_in[26];

  // workspace layout
  char* wsp = (char*)d_ws;
  auto alloc = [&](size_t bytes) -> char* {
    char* p = wsp;
    wsp += (bytes + 255) & ~(size_t)255;
    return p;
  };
  float*  hid   = (float*)  alloc((size_t)MM * DD * 4);
  __bf16* hidb  = (__bf16*) alloc((size_t)MM * DD * 2);
  __bf16* wt    = (__bf16*) alloc((size_t)DD * FFD * 2);   // reused; >= 1152*384 too
  float*  bqkv  = (float*)  alloc((size_t)NQKV * 4);
  float*  qkv   = (float*)  alloc((size_t)MM * NQKV * 4);
  float*  beta  = (float*)  alloc((size_t)MM * HH * 4);
  float*  o2    = (float*)  alloc((size_t)2 * MM * DD * 4);
  __bf16* obf   = (__bf16*) alloc((size_t)MM * DD * 2);
  float*  attn  = (float*)  alloc((size_t)MM * DD * 4);
  __bf16* ffn1b = (__bf16*) alloc((size_t)MM * FFD * 2);
  float*  ffn2  = (float*)  alloc((size_t)MM * DD * 4);

  const dim3 gemmBlk(128);
  const dim3 gridD(DD / 64,   MM / 64);   // N=384  ->  6 x 128
  const dim3 gridQ(NQKV / 64, MM / 64);   // N=1152 -> 18 x 128
  const dim3 gridF(FFD / 64,  MM / 64);   // N=1536 -> 24 x 128

  embed_ln_kernel<<<MM, 384, 0, stream>>>(ids, emb_tok, emb_pos, emb_tt,
                                          en_g, en_b, hid, hidb);

  for (int l = 0; l < LLAY; ++l) {
    const float* Wq_l = Wq + (size_t)l * DD * DD;
    const float* Wk_l = Wk + (size_t)l * DD * DD;
    const float* Wv_l = Wv + (size_t)l * DD * DD;
    const float* Wo_l = Wo + (size_t)l * DD * DD;
    const float* W1_l = W1 + (size_t)l * DD * FFD;
    const float* W2_l = W2 + (size_t)l * FFD * DD;

    // fused Q|K|V projection: pack Wt (1152 x 384) and bias, single WMMA GEMM
    conv_w_kernel<<<(DD * DD + 255) / 256, 256, 0, stream>>>(Wq_l, wt, DD, DD);
    conv_w_kernel<<<(DD * DD + 255) / 256, 256, 0, stream>>>(Wk_l, wt + (size_t)DD * DD, DD, DD);
    conv_w_kernel<<<(DD * DD + 255) / 256, 256, 0, stream>>>(Wv_l, wt + (size_t)2 * DD * DD, DD, DD);
    pack_bias_kernel<<<(NQKV + 255) / 256, 256, 0, stream>>>(
        bq + (size_t)l * DD, bk + (size_t)l * DD, bv + (size_t)l * DD, bqkv);
    gemm_bf16_kernel<0><<<gridQ, gemmBlk, 0, stream>>>(hidb, wt, bqkv, qkv, MM, NQKV, DD);

    // beta gate
    beta_kernel<<<(MM * HH + 255) / 256, 256, 0, stream>>>(
        hid, Wbeta + (size_t)l * DD * HH, bbeta + (size_t)l * HH, mask, beta);

    // delta-rule scan (fast + slow decay)
    delta_kernel<<<BB * HH, 64, 0, stream>>>(qkv, beta,
                                             fdec + (size_t)l * HH,
                                             sdec + (size_t)l * HH, o2);
    combine_o_kernel<<<(MM * DD + 255) / 256, 256, 0, stream>>>(o2, obf, MM * DD);

    // output projection + residual LN
    conv_w_kernel<<<(DD * DD + 255) / 256, 256, 0, stream>>>(Wo_l, wt, DD, DD);
    gemm_bf16_kernel<0><<<gridD, gemmBlk, 0, stream>>>(obf, wt, bo + (size_t)l * DD,
                                                       attn, MM, DD, DD);
    add_ln_kernel<<<MM, 384, 0, stream>>>(hid, attn,
                                          n1_g + (size_t)l * DD, n1_b + (size_t)l * DD, hidb);

    // FFN: GEMM1 with fused bias+GELU+bf16 epilogue (no f32 intermediate)
    conv_w_kernel<<<(DD * FFD + 255) / 256, 256, 0, stream>>>(W1_l, wt, DD, FFD);
    gemm_bf16_kernel<1><<<gridF, gemmBlk, 0, stream>>>(hidb, wt, b1 + (size_t)l * FFD,
                                                       ffn1b, MM, FFD, DD);
    conv_w_kernel<<<(DD * FFD + 255) / 256, 256, 0, stream>>>(W2_l, wt, FFD, DD);
    gemm_bf16_kernel<0><<<gridD, gemmBlk, 0, stream>>>(ffn1b, wt, b2 + (size_t)l * DD,
                                                       ffn2, MM, DD, FFD);
    add_ln_kernel<<<MM, 384, 0, stream>>>(hid, ffn2,
                                          n2_g + (size_t)l * DD, n2_b + (size_t)l * DD, hidb);
  }

  pool_kernel<<<BB, 384, 0, stream>>>(hid, mask, (float*)d_out);
}
